// EntityBoundaryLoss_23381801960011
// MI455X (gfx1250) — matrix operand order repaired and checked by reading, not compiled, not run on hardware
//
#include <hip/hip_runtime.h>
#include <hip/hip_bf16.h>
#include <cstdint>

// Problem constants (fixed by the reference).
#define Bn 128
#define Tn 512
#define Hn 1024
#define Kn 7
#define STARTT 5
#define STOPT 6
#define NEGV -10000.0f

typedef __attribute__((ext_vector_type(16))) __bf16 v16bf;
typedef __attribute__((ext_vector_type(8)))  float  v8f;
typedef __attribute__((ext_vector_type(4)))  float  v4f;

// ---------------------------------------------------------------------------
// Kernel 0: split W [7,1024] f32 into zero-padded [16,1024] bf16 hi/lo arrays.
// ---------------------------------------------------------------------------
__global__ void crf_prep_w(const float* __restrict__ W,
                           __bf16* __restrict__ Whi, __bf16* __restrict__ Wlo) {
    int idx = blockIdx.x * blockDim.x + threadIdx.x;   // 0 .. 16*1024-1
    int n = idx >> 10;
    int h = idx & 1023;
    float v = (n < Kn) ? W[n * Hn + h] : 0.0f;
    __bf16 hi = (__bf16)v;
    Whi[idx] = hi;
    Wlo[idx] = (__bf16)(v - (float)hi);
}

// ---------------------------------------------------------------------------
// Kernel 1: emissions feats[bt,k] = sum_h features[bt,h]*W[k,h] + b[k]
// One wave per 16-row tile; bf16x3 split WMMA, f32 accumulate.
// Output padded to 8 floats per row (col 7 == 0).
// ---------------------------------------------------------------------------
__global__ void crf_emit_wmma(const float* __restrict__ feat,
                              const __bf16* __restrict__ Whi,
                              const __bf16* __restrict__ Wlo,
                              const float* __restrict__ bias,
                              float* __restrict__ emis) {
    const int lane = threadIdx.x & 31;
    const int wave = (blockIdx.x * blockDim.x + threadIdx.x) >> 5;
    const int row0 = wave * 16;
    const int m  = lane & 15;       // A row / B,C column
    const int hf = lane >> 4;       // lane half selector

    const float* arow = feat + (size_t)(row0 + m) * Hn;
    // B operands: lane's column n = m; rows >= 7 are zero-padded in prep.
    const __bf16* bhip = Whi + (size_t)m * Hn;
    const __bf16* blop = Wlo + (size_t)m * Hn;

    v8f acc = {0.f, 0.f, 0.f, 0.f, 0.f, 0.f, 0.f, 0.f};

    #pragma unroll 2
    for (int h0 = 0; h0 < Hn; h0 += 32) {
        // Speculative prefetch 128 floats (four loop steps) ahead; harmless
        // (silently dropped) if it runs past the buffer on the last rows.
        __builtin_prefetch(arow + h0 + 128, 0, 1);   // global_prefetch

        // A 16x32 bf16 layout: this lane needs cols [h0+16*hf .. +15]
        // (contiguous, matching B).
        v4f a0 = *(const v4f*)(arow + h0 + 16 * hf);
        v4f a1 = *(const v4f*)(arow + h0 + 16 * hf + 4);
        v4f a2 = *(const v4f*)(arow + h0 + 16 * hf + 8);
        v4f a3 = *(const v4f*)(arow + h0 + 16 * hf + 12);

        float buf[16];
        #pragma unroll
        for (int i = 0; i < 4; ++i) {
            buf[i]      = a0[i];
            buf[4 + i]  = a1[i];
            buf[8 + i]  = a2[i];
            buf[12 + i] = a3[i];
        }
        v16bf ahi, alo;
        #pragma unroll
        for (int i = 0; i < 16; ++i) {
            __bf16 h = (__bf16)buf[i];
            ahi[i] = h;
            alo[i] = (__bf16)(buf[i] - (float)h);
        }

        // B 32x16 bf16: lane holds K = 16*hf + i, contiguous 32B load.
        v16bf bhi = *(const v16bf*)(bhip + h0 + 16 * hf);
        v16bf blo = *(const v16bf*)(blop + h0 + 16 * hf);

        acc = __builtin_amdgcn_wmma_f32_16x16x32_bf16(false, ahi, false, bhi,
                                                      (short)0, acc, false, false);
        acc = __builtin_amdgcn_wmma_f32_16x16x32_bf16(false, ahi, false, blo,
                                                      (short)0, acc, false, false);
        acc = __builtin_amdgcn_wmma_f32_16x16x32_bf16(false, alo, false, bhi,
                                                      (short)0, acc, false, false);
    }

    // C/D layout: element r -> row M = r + 8*hf, column N = m.
    if (m < 8) {
        float bv = (m < Kn) ? bias[m] : 0.0f;
        #pragma unroll
        for (int r = 0; r < 8; ++r) {
            emis[(size_t)(row0 + r + 8 * hf) * 8 + m] = acc[r] + bv;
        }
    }
}

// ---------------------------------------------------------------------------
// Kernel 2: per-batch CRF forward scan + gold score. One wave per batch.
// ---------------------------------------------------------------------------
__global__ void crf_scan(const float* __restrict__ emis,
                         const int* __restrict__ tags,
                         const int* __restrict__ lengths,
                         const float* __restrict__ trans,
                         float* __restrict__ partial) {
    const int b    = blockIdx.x;
    const int lane = threadIdx.x;           // 0..31, single wave per block
    const int len  = lengths[b];

    __shared__ float lds[64 * 8];           // 64-step emission tile

    // Per-lane transitions: lane k owns trans[k][*].
    float tr[Kn];
    if (lane < Kn) {
        #pragma unroll
        for (int p = 0; p < Kn; ++p) tr[p] = trans[lane * Kn + p];
    } else {
        #pragma unroll
        for (int p = 0; p < Kn; ++p) tr[p] = NEGV;
    }

    float fv[Kn];
    #pragma unroll
    for (int p = 0; p < Kn; ++p) fv[p] = (p == STARTT) ? 0.0f : NEGV;

    const float* frow = emis + (size_t)b * Tn * 8;

    for (int t0 = 0; t0 < len; t0 += 64) {
        // Stage 64 steps (always fully resident: t0 <= 448).
        for (int c = lane; c < 128; c += 32) {
            v4f v = *(const v4f*)(frow + (size_t)t0 * 8 + c * 4);
            *(v4f*)(&lds[c * 4]) = v;
        }
        __syncthreads();

        const int tend = (t0 + 64 < len) ? (t0 + 64) : len;
        for (int t = t0; t < tend; ++t) {
            float emit = lds[(t - t0) * 8 + (lane < Kn ? lane : 0)];
            float mx = -3.0e38f;
            #pragma unroll
            for (int p = 0; p < Kn; ++p) mx = fmaxf(mx, tr[p] + fv[p]);
            float s = 0.0f;
            #pragma unroll
            for (int p = 0; p < Kn; ++p) s += __expf(tr[p] + fv[p] - mx);
            float nv = emit + mx + __logf(s);
            // Broadcast new fv vector to every lane.
            #pragma unroll
            for (int p = 0; p < Kn; ++p) fv[p] = __shfl(nv, p, 32);
        }
        __syncthreads();
    }

    // Terminal logsumexp with trans[STOP][*] (replicated on all lanes).
    float mx = -3.0e38f;
    #pragma unroll
    for (int p = 0; p < Kn; ++p) mx = fmaxf(mx, fv[p] + trans[STOPT * Kn + p]);
    float s = 0.0f;
    #pragma unroll
    for (int p = 0; p < Kn; ++p) s += __expf(fv[p] + trans[STOPT * Kn + p] - mx);
    const float fscore = mx + __logf(s);

    // ---- gold score (lane-parallel, fixed-order reduction) ----
    const int* btags = tags + (size_t)b * Tn;
    float g = 0.0f;
    for (int t = lane; t < len; t += 32)
        g += frow[(size_t)t * 8 + btags[t]];
    for (int j = lane; j <= len; j += 32) {
        int ps = (j == 0)   ? STARTT : btags[j - 1];
        int pn = (j >= len) ? STOPT  : btags[j];
        g += trans[pn * Kn + ps];
    }
    #pragma unroll
    for (int off = 16; off > 0; off >>= 1) g += __shfl_down(g, off, 32);

    if (lane == 0) partial[b] = fscore - g;
}

// ---------------------------------------------------------------------------
// Kernel 3: deterministic reduction of 128 per-batch values.
// ---------------------------------------------------------------------------
__global__ void crf_reduce(const float* __restrict__ partial, float* __restrict__ out) {
    const int lane = threadIdx.x;           // single wave
    float s = 0.0f;
    for (int i = lane; i < Bn; i += 32) s += partial[i];
    #pragma unroll
    for (int off = 16; off > 0; off >>= 1) s += __shfl_down(s, off, 32);
    if (lane == 0) out[0] = s;
}

// ---------------------------------------------------------------------------
extern "C" void kernel_launch(void* const* d_in, const int* in_sizes, int n_in,
                              void* d_out, int out_size, void* d_ws, size_t ws_size,
                              hipStream_t stream) {
    const float* features = (const float*)d_in[0];   // [B,T,H]
    const int*   tags     = (const int*)  d_in[1];   // [B,T]
    const int*   lengths  = (const int*)  d_in[2];   // [B]
    const float* W        = (const float*)d_in[3];   // [K,H]
    const float* bias     = (const float*)d_in[4];   // [K]
    const float* trans    = (const float*)d_in[5];   // [K,K]
    float* out = (float*)d_out;

    // Workspace layout.
    char* wsb = (char*)d_ws;
    __bf16* Whi    = (__bf16*)(wsb);                       // 16*1024*2 = 32 KB
    __bf16* Wlo    = (__bf16*)(wsb + 32768);               // 32 KB
    float*  emis   = (float*)(wsb + 65536);                // 65536*8*4 = 2 MB
    float*  parts  = (float*)(wsb + 65536 + (size_t)Bn * Tn * 8 * 4);

    // 0) Split W into bf16 hi/lo, zero-padded to 16 rows.
    crf_prep_w<<<64, 256, 0, stream>>>(W, Whi, Wlo);

    // 1) Emission GEMM: 4096 waves of 16 rows each (8 waves / 256-thr block).
    crf_emit_wmma<<<(Bn * Tn / 16) / 8, 256, 0, stream>>>(features, Whi, Wlo, bias, emis);

    // 2) Forward scan + gold score: one wave per batch.
    crf_scan<<<Bn, 32, 0, stream>>>(emis, tags, lengths, trans, parts);

    // 3) Deterministic final reduction.
    crf_reduce<<<1, 32, 0, stream>>>(parts, out);
}
